// ComplexPolarTransformerBeta_31069793419830
// MI455X (gfx1250) — matrix operand order repaired and checked by reading, not compiled, not run on hardware
//
#include <hip/hip_runtime.h>
#include <hip/hip_bf16.h>
#include <math.h>

#define NN 16384      // NUM_NODES
#define NE 131072     // NUM_EDGES
#define NM 512        // NUM_MOLS
#define HD 256        // hidden dim
#define NL 3          // layers
#define NRBF 50
#define PI_F 3.14159265358979f

typedef __attribute__((ext_vector_type(16))) _Float16 v16h;
typedef __attribute__((ext_vector_type(8)))  float    v8f;

// ---------------------------------------------------------------------------
// Transpose-convert weights:  Bt[n][k] (f16, col-major of B) = B[k][n] (f32)
// Makes each lane's WMMA B-fragment 16 *contiguous* halves -> 2x global_load_b128.
// ---------------------------------------------------------------------------
__global__ __launch_bounds__(256) void convB_kernel(const float* __restrict__ B,
                                                    _Float16* __restrict__ Bt,
                                                    int K, int N) {
  int idx = blockIdx.x * blockDim.x + threadIdx.x;
  if (idx >= K * N) return;
  int k = idx / N, n = idx - k * N;
  Bt[(size_t)n * K + k] = (_Float16)B[idx];
}

// ---------------------------------------------------------------------------
// Generic WMMA GEMM:  C[M,N] = (HAS_ADD? addsrc : 0) + f16(A[M,K]) @ Bt^T + bias
// A: f32 row-major [M,K]; Bt: f16 col-major [N,K] (from convB_kernel).
// One wave32 per 16x16 output tile, K stepped by 32 via v_wmma_f32_16x16x32_f16.
// M%16==0, N%16==0, K%128==0 (K loop unrolled x4).
// Fragment layouts used here (contiguous k = 16*half + s in A and B):
//   A (16x32): lanes 0-15 row M=lane, elems K0..15; lanes 16-31 K16..31
//   B (32x16): lanes 0-15 col N=lane carry K0..15; lanes 16-31 carry K16..31
//   C (8 VGPR): row = r + 8*(lane>=16), col = lane&15
// ---------------------------------------------------------------------------
template <bool HAS_BIAS, bool HAS_ADD>
__global__ __launch_bounds__(256) void gemm_wmma_f16(
    const float* __restrict__ A, const _Float16* __restrict__ Bt,
    const float* __restrict__ bias, const float* __restrict__ addsrc,
    float* __restrict__ C, int M, int N, int K)
{
  const int wave = threadIdx.x >> 5;
  const int lane = threadIdx.x & 31;
  const int tilesN = N >> 4;
  const int w = blockIdx.x * 8 + wave;
  const int total = (M >> 4) * tilesN;
  if (w >= total) return;                 // wave-uniform: EXEC stays all-ones
  const int tM = w / tilesN, tN = w % tilesN;
  const int half = lane >> 4;             // 0: lanes 0-15, 1: lanes 16-31
  const int li   = lane & 15;
  const int rowA = tM * 16 + li;
  const int colB = tN * 16 + li;

  const float*    Arow  = A  + (size_t)rowA * K + half * 16;  // 32B-aligned
  const _Float16* Btrow = Bt + (size_t)colB * K + half * 16;  // 32B-aligned
  v8f acc = {};
#pragma unroll 4
  for (int k0 = 0; k0 < K; k0 += 32) {
    // A fragment: two aligned float4 pairs -> global_load_b128 + v_cvt_pk_f16_f32
    const float4* ap = (const float4*)(Arow + k0);
    float4 a0 = ap[0], a1 = ap[1];        // K = k0+half*16 + 0..7
    float4 a2 = ap[2], a3 = ap[3];        // K = k0+half*16 + 8..15
    __builtin_prefetch(Arow + k0 + 32, 0, 3);
    v16h a;
    a[0]=(_Float16)a0.x; a[1]=(_Float16)a0.y; a[2]=(_Float16)a0.z; a[3]=(_Float16)a0.w;
    a[4]=(_Float16)a1.x; a[5]=(_Float16)a1.y; a[6]=(_Float16)a1.z; a[7]=(_Float16)a1.w;
    a[8]=(_Float16)a2.x; a[9]=(_Float16)a2.y; a[10]=(_Float16)a2.z; a[11]=(_Float16)a2.w;
    a[12]=(_Float16)a3.x; a[13]=(_Float16)a3.y; a[14]=(_Float16)a3.z; a[15]=(_Float16)a3.w;
    // B fragment: 16 contiguous f16 (32B) -> two global_load_b128, linear address
    v16h b = *(const v16h*)(Btrow + k0);

    acc = __builtin_amdgcn_wmma_f32_16x16x32_f16(
        /*neg_a=*/false, a, /*neg_b=*/false, b,
        /*c_mod=*/(short)0, acc, /*reuse_a=*/false, /*reuse_b=*/false);
  }
  float bv = 0.0f;
  if (HAS_BIAS) bv = bias[colB];
#pragma unroll
  for (int r = 0; r < 8; ++r) {
    const int row = tM * 16 + r + half * 8;
    const size_t idx = (size_t)row * N + colB;
    float o = acc[r] + bv;
    if (HAS_ADD) o += addsrc[idx];
    C[idx] = o;
  }
}

// ---------------------------------------------------------------------------
__global__ __launch_bounds__(256) void zero_f32(float* p, int n) {
  int i = blockIdx.x * blockDim.x + threadIdx.x;
  if (i < n) p[i] = 0.0f;
}

__global__ __launch_bounds__(256) void rbf_kernel(const float* __restrict__ dist,
                                                  float* __restrict__ rbf) {
  int idx = blockIdx.x * blockDim.x + threadIdx.x;
  if (idx >= NE * NRBF) return;
  int e = idx / NRBF, j = idx - e * NRBF;
  float d = dist[e];
  float c = j * (5.0f / 49.0f);                 // linspace(0, 5, 50)
  float env = (d < 5.0f) ? 0.5f * (cosf(PI_F * d * 0.2f) + 1.0f) : 0.0f;
  float u = (d - c) * 10.0f;                    // delta = 5/50 = 0.1
  rbf[idx] = expf(-0.5f * u * u) * env;
}

__global__ __launch_bounds__(256) void embed_kernel(
    const float* __restrict__ at, const float* __restrict__ cs,
    const float* __restrict__ Wm, const float* __restrict__ bm,
    const float* __restrict__ Wp, const float* __restrict__ bp,
    float* __restrict__ mag, float* __restrict__ ph)
{
  int n = blockIdx.x, h = threadIdx.x;
  float x[8];
#pragma unroll
  for (int d = 0; d < 5; ++d) x[d] = at[n * 5 + d];
#pragma unroll
  for (int d = 0; d < 3; ++d) x[5 + d] = cs[n * 3 + d];
  float zm = bm[h], zp = bp[h];
#pragma unroll
  for (int d = 0; d < 8; ++d) {
    zm += x[d] * Wm[d * HD + h];
    zp += x[d] * Wp[d * HD + h];
  }
  mag[(size_t)n * HD + h] = (zm > 20.0f) ? zm : log1pf(expf(zm));
  ph[(size_t)n * HD + h]  = PI_F * tanhf(zp);
}

__global__ __launch_bounds__(256) void attn_init(unsigned* nmax, float* nsum, int n) {
  int i = blockIdx.x * blockDim.x + threadIdx.x;
  if (i < n) { nmax[i] = 0x007FFFFFu; /* enc(-inf) */ nsum[i] = 0.0f; }
}

// one wave32 per edge: score = sum_h qm[d]km[s]cos(qp[d]-kp[s]) / 16 + rbf.We + be
__global__ __launch_bounds__(256) void score_kernel(
    const int* __restrict__ ei, const float* __restrict__ qm,
    const float* __restrict__ km, const float* __restrict__ qp,
    const float* __restrict__ kp, const float* __restrict__ rbf,
    const float* __restrict__ We, const float* __restrict__ be,
    float* __restrict__ score, unsigned* __restrict__ nmax)
{
  int gw = (blockIdx.x * blockDim.x + threadIdx.x) >> 5;
  int lane = threadIdx.x & 31;
  if (gw >= NE) return;
  int s = ei[gw], d = ei[NE + gw];
  const float* qmd = qm + (size_t)d * HD; const float* kms = km + (size_t)s * HD;
  const float* qpd = qp + (size_t)d * HD; const float* kps = kp + (size_t)s * HD;
  float part = 0.0f;
#pragma unroll
  for (int j = 0; j < 8; ++j) {
    int h = lane + 32 * j;
    part += qmd[h] * kms[h] * cosf(qpd[h] - kps[h]);
  }
  part *= 0.0625f;                         // 1/sqrt(256)
  const float* re = rbf + (size_t)gw * NRBF;
  part += re[lane] * We[lane];             // lane < 32 < 50 always valid
  if (lane < NRBF - 32) part += re[lane + 32] * We[lane + 32];
#pragma unroll
  for (int m = 16; m; m >>= 1) part += __shfl_xor(part, m, 32);
  if (lane == 0) {
    float sc = part + be[0];
    score[gw] = sc;
    unsigned u = __float_as_uint(sc);
    u = (u & 0x80000000u) ? ~u : (u | 0x80000000u);   // order-preserving encode
    atomicMax(nmax + d, u);
  }
}

__global__ __launch_bounds__(256) void expsum_kernel(
    const int* __restrict__ ei, const float* __restrict__ score,
    const unsigned* __restrict__ nmax, float* __restrict__ eexp,
    float* __restrict__ nsum)
{
  int e = blockIdx.x * blockDim.x + threadIdx.x;
  if (e >= NE) return;
  int d = ei[NE + e];
  unsigned u = nmax[d];
  float m = (u & 0x80000000u) ? __uint_as_float(u ^ 0x80000000u) : __uint_as_float(~u);
  float ee = expf(score[e] - m);
  eexp[e] = ee;
  atomicAdd(nsum + d, ee);
}

// one wave32 per edge: new_mag[dst] += alpha*vm[src]; new_ph[dst] += alpha*vp[src]
__global__ __launch_bounds__(256) void attn_scatter(
    const int* __restrict__ ei, const float* __restrict__ eexp,
    const float* __restrict__ nsum, const float* __restrict__ vm,
    const float* __restrict__ vp, float* __restrict__ nm, float* __restrict__ np)
{
  int gw = (blockIdx.x * blockDim.x + threadIdx.x) >> 5;
  int lane = threadIdx.x & 31;
  if (gw >= NE) return;
  int s = ei[gw], d = ei[NE + gw];
  float alpha = eexp[gw] / (nsum[d] + 1e-9f);
#pragma unroll
  for (int j = 0; j < 8; ++j) {
    int h = lane + 32 * j;
    atomicAdd(nm + (size_t)d * HD + h, alpha * vm[(size_t)s * HD + h]);
    atomicAdd(np + (size_t)d * HD + h, alpha * vp[(size_t)s * HD + h]);
  }
}

// fused gate GEMM (K=50, weights cache-resident) + scatter: avoids [E,256] gate tensor
__global__ __launch_bounds__(256) void gate_scatter(
    const int* __restrict__ ei, const float* __restrict__ rbf,
    const float* __restrict__ We, const float* __restrict__ be,
    const float* __restrict__ nm, const float* __restrict__ np,
    float* __restrict__ am, float* __restrict__ ap_)
{
  int gw = (blockIdx.x * blockDim.x + threadIdx.x) >> 5;
  int lane = threadIdx.x & 31;
  if (gw >= NE) return;
  int s = ei[gw], d = ei[NE + gw];
  const float* re = rbf + (size_t)gw * NRBF;
  float g[8];
#pragma unroll
  for (int j = 0; j < 8; ++j) g[j] = be[lane + 32 * j];
  for (int k = 0; k < NRBF; ++k) {
    float rv = re[k];
    const float* wrow = We + (size_t)k * HD;
#pragma unroll
    for (int j = 0; j < 8; ++j) g[j] += rv * wrow[lane + 32 * j];
  }
#pragma unroll
  for (int j = 0; j < 8; ++j) {
    float z = g[j];
    float gg = z / (1.0f + expf(-z));      // silu
    int h = lane + 32 * j;
    atomicAdd(am  + (size_t)d * HD + h, gg * nm[(size_t)s * HD + h]);
    atomicAdd(ap_ + (size_t)d * HD + h, gg * np[(size_t)s * HD + h]);
  }
}

// residual + layernorm(mag); ph gets plain residual
__global__ __launch_bounds__(256) void ln_kernel(
    const float* __restrict__ gm, const float* __restrict__ gp,
    const float* __restrict__ g, const float* __restrict__ b,
    float* __restrict__ mag, float* __restrict__ ph)
{
  __shared__ float red[256];
  int n = blockIdx.x, h = threadIdx.x;
  size_t idx = (size_t)n * HD + h;
  float rm = gm[idx] + mag[idx];
  float rp = gp[idx] + ph[idx];
  red[h] = rm; __syncthreads();
  for (int s2 = 128; s2; s2 >>= 1) { if (h < s2) red[h] += red[h + s2]; __syncthreads(); }
  float mu = red[0] / HD; __syncthreads();
  float dv = rm - mu; red[h] = dv * dv; __syncthreads();
  for (int s2 = 128; s2; s2 >>= 1) { if (h < s2) red[h] += red[h + s2]; __syncthreads(); }
  float var = red[0] / HD;
  mag[idx] = (rm - mu) * rsqrtf(var + 1e-5f) * g[h] + b[h];
  ph[idx] = rp;
}

__global__ __launch_bounds__(256) void trep_kernel(const float* __restrict__ mag,
                                                   const float* __restrict__ ph,
                                                   float* __restrict__ t) {
  int i = blockIdx.x * blockDim.x + threadIdx.x;
  if (i < NN * HD) t[i] = mag[i] * cosf(ph[i]);
}

__global__ __launch_bounds__(256) void pool_kernel(const float* __restrict__ ar,
                                                   const int* __restrict__ n2m,
                                                   float* __restrict__ s,
                                                   float* __restrict__ cnt) {
  int idx = blockIdx.x * blockDim.x + threadIdx.x;
  if (idx >= NN * HD) return;
  int n = idx >> 8, h = idx & 255;
  int m = n2m[n];
  atomicAdd(s + (size_t)m * HD + h, ar[idx]);
  if (h == 0) atomicAdd(cnt + m, 1.0f);
}

__global__ __launch_bounds__(256) void head1_kernel(
    const float* __restrict__ s, const float* __restrict__ cnt,
    const float* __restrict__ W1, const float* __restrict__ b1, float* __restrict__ h1) {
  int idx = blockIdx.x * blockDim.x + threadIdx.x;
  if (idx >= NM * HD) return;
  int b = idx >> 8, j = idx & 255;
  float c = cnt[b]; c = c > 1.0f ? c : 1.0f;
  float inv = 1.0f / c;
  float acc = b1[j];
  const float* sb = s + (size_t)b * HD;
  for (int i = 0; i < HD; ++i) {
    float sv = sb[i];
    acc += sv * inv * W1[(size_t)i * HD + j];          // mean half
    acc += sv * W1[(size_t)(HD + i) * HD + j];         // sum half
  }
  h1[idx] = acc / (1.0f + expf(-acc));
}

__global__ __launch_bounds__(256) void head2_kernel(
    const float* __restrict__ h1, const float* __restrict__ W2,
    const float* __restrict__ b2, float* __restrict__ h2) {
  int idx = blockIdx.x * blockDim.x + threadIdx.x;
  if (idx >= NM * 128) return;
  int b = idx >> 7, j = idx & 127;
  float acc = b2[j];
  const float* hb = h1 + (size_t)b * HD;
  for (int i = 0; i < HD; ++i) acc += hb[i] * W2[(size_t)i * 128 + j];
  h2[idx] = acc / (1.0f + expf(-acc));
}

__global__ __launch_bounds__(256) void head3_kernel(
    const float* __restrict__ h2, const float* __restrict__ W3,
    const float* __restrict__ b3, float* __restrict__ out) {
  int b = blockIdx.x * blockDim.x + threadIdx.x;
  if (b >= NM) return;
  float acc = b3[0];
  const float* hb = h2 + (size_t)b * 128;
  for (int i = 0; i < 128; ++i) acc += hb[i] * W3[i];
  out[b] = acc;
}

// ---------------------------------------------------------------------------
extern "C" void kernel_launch(void* const* d_in, const int* in_sizes, int n_in,
                              void* d_out, int out_size, void* d_ws, size_t ws_size,
                              hipStream_t stream) {
  (void)in_sizes; (void)n_in; (void)out_size; (void)ws_size;
  const float* atom_types = (const float*)d_in[0];
  const float* coords     = (const float*)d_in[1];
  const float* edge_dist  = (const float*)d_in[2];
  const int*   edge_index = (const int*)d_in[3];
  const int*   node2mol   = (const int*)d_in[4];
  const float* emb_Wm = (const float*)d_in[5];
  const float* emb_bm = (const float*)d_in[6];
  const float* emb_Wp = (const float*)d_in[7];
  const float* emb_bp = (const float*)d_in[8];
  const float* attn_Wqm = (const float*)d_in[9];
  const float* attn_Wkm = (const float*)d_in[10];
  const float* attn_Wvm = (const float*)d_in[11];
  const float* attn_Wqp = (const float*)d_in[12];
  const float* attn_Wkp = (const float*)d_in[13];
  const float* attn_Wvp = (const float*)d_in[14];
  const float* attn_We  = (const float*)d_in[15];
  const float* attn_be  = (const float*)d_in[16];
  const float* mp_We = (const float*)d_in[17];
  const float* mp_be = (const float*)d_in[18];
  const float* mp_Wm = (const float*)d_in[19];
  const float* mp_bm = (const float*)d_in[20];
  const float* mp_Wp = (const float*)d_in[21];
  const float* mp_bp = (const float*)d_in[22];
  const float* ln_g = (const float*)d_in[23];
  const float* ln_b = (const float*)d_in[24];
  const float* proj_W = (const float*)d_in[25];
  const float* proj_b = (const float*)d_in[26];
  const float* head_W1 = (const float*)d_in[27];
  const float* head_b1 = (const float*)d_in[28];
  const float* head_W2 = (const float*)d_in[29];
  const float* head_b2 = (const float*)d_in[30];
  const float* head_W3 = (const float*)d_in[31];
  const float* head_b3 = (const float*)d_in[32];
  float* out = (float*)d_out;

  // workspace layout (f32 units; every slab is a multiple of 8 floats -> 32B aligned)
  float* wsf = (float*)d_ws;
  size_t o = 0;
  const size_t NHsz = (size_t)NN * HD;
  float* rbf  = wsf + o; o += (size_t)NE * NRBF;
  float* mag  = wsf + o; o += NHsz;
  float* ph   = wsf + o; o += NHsz;
  float* bA   = wsf + o; o += NHsz;   // qm -> agg_m -> t
  float* bB   = wsf + o; o += NHsz;   // km -> gemm_m out -> atom_repr
  float* bC   = wsf + o; o += NHsz;   // vm
  float* bD   = wsf + o; o += NHsz;   // qp -> agg_p
  float* bE   = wsf + o; o += NHsz;   // kp -> gemm_p out
  float* bF   = wsf + o; o += NHsz;   // vp
  float* bG   = wsf + o; o += NHsz;   // new_mag (attention)   \ contiguous pair
  float* bH   = wsf + o; o += NHsz;   // new_ph  (attention)   /
  float* score = wsf + o; o += NE;
  float* eexp  = wsf + o; o += NE;
  float* nsum  = wsf + o; o += NN;
  unsigned* nmax = (unsigned*)(wsf + o); o += NN;
  float* smol = wsf + o; o += (size_t)NM * HD;   // \ contiguous with cnt
  float* cnt  = wsf + o; o += NM;                // /
  float* h1   = wsf + o; o += (size_t)NM * HD;
  float* h2   = wsf + o; o += (size_t)NM * 128;
  _Float16* Bt = (_Float16*)(wsf + o); o += (size_t)HD * HD / 2;  // f16 transposed weight

  const int T = 256;
  const int gemmBlocks = ((NN / 16) * (HD / 16)) / 8;     // 2048
  const int convBlocks = (HD * HD + T - 1) / T;           // 256
  const int edgeWarpBlocks = (NE * 32) / T;               // 16384
  const int flatNH = (int)(NHsz);

  // RBF + embedding
  rbf_kernel<<<(NE * NRBF + T - 1) / T, T, 0, stream>>>(edge_dist, rbf);
  embed_kernel<<<NN, T, 0, stream>>>(atom_types, coords, emb_Wm, emb_bm, emb_Wp, emb_bp, mag, ph);

  for (int l = 0; l < NL; ++l) {
    const float* Wqm = attn_Wqm + (size_t)l * HD * HD;
    const float* Wkm = attn_Wkm + (size_t)l * HD * HD;
    const float* Wvm = attn_Wvm + (size_t)l * HD * HD;
    const float* Wqp = attn_Wqp + (size_t)l * HD * HD;
    const float* Wkp = attn_Wkp + (size_t)l * HD * HD;
    const float* Wvp = attn_Wvp + (size_t)l * HD * HD;

    // QKV projections (WMMA, no bias / no add); Bt converted just-in-time, stream-ordered
    convB_kernel<<<convBlocks, T, 0, stream>>>(Wqm, Bt, HD, HD);
    gemm_wmma_f16<false, false><<<gemmBlocks, T, 0, stream>>>(mag, Bt, nullptr, nullptr, bA, NN, HD, HD);
    convB_kernel<<<convBlocks, T, 0, stream>>>(Wkm, Bt, HD, HD);
    gemm_wmma_f16<false, false><<<gemmBlocks, T, 0, stream>>>(mag, Bt, nullptr, nullptr, bB, NN, HD, HD);
    convB_kernel<<<convBlocks, T, 0, stream>>>(Wvm, Bt, HD, HD);
    gemm_wmma_f16<false, false><<<gemmBlocks, T, 0, stream>>>(mag, Bt, nullptr, nullptr, bC, NN, HD, HD);
    convB_kernel<<<convBlocks, T, 0, stream>>>(Wqp, Bt, HD, HD);
    gemm_wmma_f16<false, false><<<gemmBlocks, T, 0, stream>>>(ph,  Bt, nullptr, nullptr, bD, NN, HD, HD);
    convB_kernel<<<convBlocks, T, 0, stream>>>(Wkp, Bt, HD, HD);
    gemm_wmma_f16<false, false><<<gemmBlocks, T, 0, stream>>>(ph,  Bt, nullptr, nullptr, bE, NN, HD, HD);
    convB_kernel<<<convBlocks, T, 0, stream>>>(Wvp, Bt, HD, HD);
    gemm_wmma_f16<false, false><<<gemmBlocks, T, 0, stream>>>(ph,  Bt, nullptr, nullptr, bF, NN, HD, HD);

    // attention scores + segment softmax
    attn_init<<<(NN + T - 1) / T, T, 0, stream>>>(nmax, nsum, NN);
    score_kernel<<<edgeWarpBlocks, T, 0, stream>>>(edge_index, bA, bB, bD, bE, rbf,
                                                   attn_We + (size_t)l * NRBF, attn_be + l,
                                                   score, nmax);
    expsum_kernel<<<(NE + T - 1) / T, T, 0, stream>>>(edge_index, score, nmax, eexp, nsum);

    // alpha-weighted value scatter
    zero_f32<<<(2 * flatNH + T - 1) / T, T, 0, stream>>>(bG, 2 * flatNH);
    attn_scatter<<<edgeWarpBlocks, T, 0, stream>>>(edge_index, eexp, nsum, bC, bF, bG, bH);

    // gated message passing scatter (fused gate GEMM, K=50)
    zero_f32<<<(flatNH + T - 1) / T, T, 0, stream>>>(bA, flatNH);
    zero_f32<<<(flatNH + T - 1) / T, T, 0, stream>>>(bD, flatNH);
    gate_scatter<<<edgeWarpBlocks, T, 0, stream>>>(edge_index, rbf,
                                                   mp_We + (size_t)l * NRBF * HD,
                                                   mp_be + (size_t)l * HD,
                                                   bG, bH, bA, bD);

    // new = new_attn + agg @ W + b   (WMMA with bias + addsrc, branch-free epilogue)
    convB_kernel<<<convBlocks, T, 0, stream>>>(mp_Wm + (size_t)l * HD * HD, Bt, HD, HD);
    gemm_wmma_f16<true, true><<<gemmBlocks, T, 0, stream>>>(bA, Bt,
                                                mp_bm + (size_t)l * HD, bG, bB, NN, HD, HD);
    convB_kernel<<<convBlocks, T, 0, stream>>>(mp_Wp + (size_t)l * HD * HD, Bt, HD, HD);
    gemm_wmma_f16<true, true><<<gemmBlocks, T, 0, stream>>>(bD, Bt,
                                                mp_bp + (size_t)l * HD, bH, bE, NN, HD, HD);

    // residual + layernorm
    ln_kernel<<<NN, T, 0, stream>>>(bB, bE, ln_g + (size_t)l * HD, ln_b + (size_t)l * HD, mag, ph);
  }

  // atom_repr = (mag * cos(ph)) @ proj_W + proj_b
  trep_kernel<<<(flatNH + T - 1) / T, T, 0, stream>>>(mag, ph, bA);
  convB_kernel<<<convBlocks, T, 0, stream>>>(proj_W, Bt, HD, HD);
  gemm_wmma_f16<true, false><<<gemmBlocks, T, 0, stream>>>(bA, Bt, proj_b, nullptr, bB, NN, HD, HD);

  // pooling + head
  zero_f32<<<(NM * HD + NM + T - 1) / T, T, 0, stream>>>(smol, NM * HD + NM);
  pool_kernel<<<(flatNH + T - 1) / T, T, 0, stream>>>(bB, node2mol, smol, cnt);
  head1_kernel<<<(NM * HD + T - 1) / T, T, 0, stream>>>(smol, cnt, head_W1, head_b1, h1);
  head2_kernel<<<(NM * 128 + T - 1) / T, T, 0, stream>>>(h1, head_W2, head_b2, h2);
  head3_kernel<<<(NM + T - 1) / T, T, 0, stream>>>(h2, head_W3, head_b3, out);
}
